// LocalAggregator_72129680769655
// MI455X (gfx1250) — matrix operand order repaired and checked by reading, not compiled, not run on hardware
//
#include <hip/hip_runtime.h>
#include <hip/hip_bf16.h>

#define BATCH 256
#define NN 256
#define DD 128

typedef __bf16 bf16;
typedef __attribute__((ext_vector_type(16))) __bf16 v16bf;
typedef __attribute__((ext_vector_type(8)))  __bf16 v8bf;
typedef __attribute__((ext_vector_type(4)))  __bf16 v4bf;
typedef __attribute__((ext_vector_type(8)))  float  v8f;
typedef __attribute__((ext_vector_type(8)))  unsigned v8u;

// LDS geometry (element strides). Rows padded so 16-lane tile accesses are bank-conflict-free.
#define SRC_STRIDE   136   // bf16: 128 + 8
#define T_STRIDE     264   // bf16: 256 + 8
#define A_STRIDE     136
#define SC_STRIDE    260   // f32: 256 cols + slots: [256]=rowmax, [257]=inv-sum
#define SC_WAVE      (16 * SC_STRIDE)
#define SMEM_BYTES   ((NN*SRC_STRIDE + DD*T_STRIDE + 4*A_STRIDE) * 2 + 8 * SC_WAVE * 4)

__device__ __forceinline__ v16bf ldfrag(const bf16* p0, const bf16* p1) {
  v8bf lo = *(const v8bf*)p0;
  v8bf hi = *(const v8bf*)p1;
  v16bf r;
#pragma unroll
  for (int i = 0; i < 8; ++i) { r[i] = lo[i]; r[i + 8] = hi[i]; }
  return r;
}

// packed bf16 multiply of two 16-element fragments (8x v_pk_mul_bf16)
__device__ __forceinline__ v16bf pkmul(v16bf a, v16bf s) {
  v8u ua = __builtin_bit_cast(v8u, a);
  v8u us = __builtin_bit_cast(v8u, s);
  v8u ut;
#pragma unroll
  for (int h = 0; h < 8; ++h) {
    unsigned t;
    asm("v_pk_mul_bf16 %0, %1, %2" : "=v"(t) : "v"(ua[h]), "v"(us[h]));
    ut[h] = t;
  }
  return __builtin_bit_cast(v16bf, ut);
}

__global__ __launch_bounds__(256, 1)
void LocalAggregator_gat_kernel(const float* __restrict__ src,
                                const int*   __restrict__ adj,
                                const float* __restrict__ a0,
                                const float* __restrict__ a1,
                                const float* __restrict__ a2,
                                const float* __restrict__ a3,
                                const float* __restrict__ bias,
                                float*       __restrict__ out)
{
  extern __shared__ char smem[];
  bf16* sh_src  = (bf16*)smem;                  // [256][136] row-major src (bf16)
  bf16* sh_srcT = sh_src  + NN*SRC_STRIDE;      // [128][264] transposed src
  bf16* sh_a    = sh_srcT + DD*T_STRIDE;        // [4][136]   a_k (bf16)
  float* sh_sc  = (float*)(sh_a + 4*A_STRIDE);  // [8 waves][16][260] f32 scores / packed bf16 e (in place)

  const int b    = blockIdx.x;
  const int tid  = threadIdx.x;
  const int lane = tid & 31;
  const int wv   = tid >> 5;
  const int half = lane >> 4;    // 0: lanes 0-15, 1: lanes 16-31
  const int l16  = lane & 15;
  const int c8   = half * 8;     // A-fragment K-chunk base
  const int c16  = half * 16;    // unused; B fragments also use c8

  // ---------- Phase 0: stage src[b] (both layouts) and a_k into LDS ----------
  {
    const unsigned soff = (unsigned)(b * NN + tid) * DD;     // 32-bit offsets (all tensors < 2^31)
    const float* srow = src + soff;
    bf16* dr = sh_src + tid * SRC_STRIDE;
#pragma unroll
    for (int q = 0; q < DD / 4; ++q) {
      float4 v = ((const float4*)srow)[q];
      v4bf h; h[0] = (bf16)v.x; h[1] = (bf16)v.y; h[2] = (bf16)v.z; h[3] = (bf16)v.w;
      *(v4bf*)(dr + q * 4) = h;
      const int d = q * 4;
      sh_srcT[(d + 0) * T_STRIDE + tid] = h[0];
      sh_srcT[(d + 1) * T_STRIDE + tid] = h[1];
      sh_srcT[(d + 2) * T_STRIDE + tid] = h[2];
      sh_srcT[(d + 3) * T_STRIDE + tid] = h[3];
    }
    if (tid < DD) {
      sh_a[0 * A_STRIDE + tid] = (bf16)a0[tid];
      sh_a[1 * A_STRIDE + tid] = (bf16)a1[tid];
      sh_a[2 * A_STRIDE + tid] = (bf16)a2[tid];
      sh_a[3 * A_STRIDE + tid] = (bf16)a3[tid];
    }
  }
  __syncthreads();

  float* scw = sh_sc + wv * SC_WAVE;            // this wave's score block

#pragma unroll 1
  for (int it = 0; it < 2; ++it) {
    const int itile = wv * 2 + it;              // 8 waves x 2 = 16 row-tiles
    const int rowA  = itile * 16 + l16;

    // ----- pre-scaled A fragments: Ak[ktile][k] = src_row (.) a_k (packed bf16 muls) -----
    v16bf Ak[4][4];
#pragma unroll
    for (int kt = 0; kt < 4; ++kt) {
      const bf16* ar = sh_src + rowA * SRC_STRIDE + kt * 32;
      v16bf a = ldfrag(ar + c8, ar + 16 + c8);
#pragma unroll
      for (int k = 0; k < 4; ++k) {
        const bf16* sp = sh_a + k * A_STRIDE + kt * 32;
        Ak[kt][k] = pkmul(a, ldfrag(sp + c8, sp + 16 + c8));
      }
    }

    // ----- pass 1: scores (4 WMMAs per k-step), fused adj-select + lrelu -> LDS f32 -----
    float mx[8];
#pragma unroll
    for (int r = 0; r < 8; ++r) mx[r] = -3.402823466e38f;

    const unsigned abase = (unsigned)(b * NN + itile * 16 + half * 8) * NN + l16;

#pragma unroll
    for (int jt = 0; jt < 16; ++jt) {
      v8f e0 = {}, e1 = {}, e2 = {}, e3 = {};
#pragma unroll
      for (int kt = 0; kt < 4; ++kt) {
        const bf16* br = sh_src + (jt * 16 + l16) * SRC_STRIDE + kt * 32 + c8;
        v16bf bf = ldfrag(br, br + 16);
        e0 = __builtin_amdgcn_wmma_f32_16x16x32_bf16(false, Ak[kt][0], false, bf, (short)0, e0, false, false);
        e1 = __builtin_amdgcn_wmma_f32_16x16x32_bf16(false, Ak[kt][1], false, bf, (short)0, e1, false, false);
        e2 = __builtin_amdgcn_wmma_f32_16x16x32_bf16(false, Ak[kt][2], false, bf, (short)0, e2, false, false);
        e3 = __builtin_amdgcn_wmma_f32_16x16x32_bf16(false, Ak[kt][3], false, bf, (short)0, e3, false, false);
      }
      const int* ap = adj + (abase + jt * 16);
#pragma unroll
      for (int r = 0; r < 8; ++r) {
        const int av = ap[r * NN];
        const float e01 = (av == 1) ? e0[r] : e1[r];
        const float e23 = (av == 3) ? e2[r] : e3[r];
        float e = (av < 3) ? e01 : e23;
        e = fmaxf(e, 0.2f * e);                              // leaky relu
        const float v = ((unsigned)(av - 1) < 4u) ? e : -3.402823466e38f;
        scw[(r + half * 8) * SC_STRIDE + jt * 16 + l16] = v;
        mx[r] = fmaxf(mx[r], v);
      }
    }

    // ----- row max across the 16-lane half-groups; stash into slot [256] -----
#pragma unroll
    for (int r = 0; r < 8; ++r) {
      float m = mx[r];
      m = fmaxf(m, __shfl_xor(m, 1, 32));
      m = fmaxf(m, __shfl_xor(m, 2, 32));
      m = fmaxf(m, __shfl_xor(m, 4, 32));
      m = fmaxf(m, __shfl_xor(m, 8, 32));
      mx[r] = m;
    }
    if (l16 == 0) {
#pragma unroll
      for (int r = 0; r < 8; ++r) scw[(r + half * 8) * SC_STRIDE + 256] = mx[r];
    }
    asm volatile("s_wait_dscnt 0" ::: "memory");

    // ----- pass 2 (row per lane-pair): e = exp(x-max), packed bf16 in place, row sum -----
    // lane owns row = lane>>1, col half = (lane&1)*128.
    // bf16 e layout per row: cols 0-127 -> bytes [0,256), cols 128-255 -> bytes [512,768)
    // (each lane writes only bytes it has already consumed -> in-place safe)
    {
      const int   prow = lane >> 1;
      const int   pcol = (lane & 1) * 128;
      float*      rrow = scw + prow * SC_STRIDE;
      bf16*       brow = (bf16*)rrow;
      const float pm   = rrow[256];
      float psum = 0.0f;
#pragma unroll
      for (int q = 0; q < 32; ++q) {
        const float4 x = *(const float4*)(rrow + pcol + q * 4);
        const float f0 = __expf(x.x - pm);
        const float f1 = __expf(x.y - pm);
        const float f2 = __expf(x.z - pm);
        const float f3 = __expf(x.w - pm);
        psum += (f0 + f1) + (f2 + f3);
        v4bf h; h[0] = (bf16)f0; h[1] = (bf16)f1; h[2] = (bf16)f2; h[3] = (bf16)f3;
        *(v4bf*)(brow + (lane & 1) * 256 + q * 4) = h;
      }
      psum += __shfl_xor(psum, 1, 32);
      if ((lane & 1) == 0) rrow[257] = 1.0f / psum;
    }
    asm volatile("s_wait_dscnt 0" ::: "memory");

    // ----- out_tile = (e @ src) * inv + bias  (K = 256 over j, B from transposed src) -----
    v16bf A2[8];
    const bf16* arow = (const bf16*)(scw + l16 * SC_STRIDE);
#pragma unroll
    for (int kk = 0; kk < 8; ++kk) {
      const int jb = kk * 32 + c8 + ((kk >= 4) ? 128 : 0);   // split-row remap (never straddles)
      A2[kk] = ldfrag(arow + jb, arow + jb + 16);
    }
    float invr[8];
#pragma unroll
    for (int r = 0; r < 8; ++r) invr[r] = scw[(r + half * 8) * SC_STRIDE + 257];

    const unsigned obase = (unsigned)(b * NN + itile * 16 + half * 8) * DD + l16;

#pragma unroll
    for (int dt = 0; dt < 8; ++dt) {
      v8f acc = {};
#pragma unroll
      for (int kk = 0; kk < 8; ++kk) {
        const bf16* p = sh_srcT + (dt * 16 + l16) * T_STRIDE + kk * 32 + c8;
        v16bf bf = ldfrag(p, p + 16);
        acc = __builtin_amdgcn_wmma_f32_16x16x32_bf16(false, A2[kk], false, bf, (short)0, acc, false, false);
      }
      const float bv = bias[dt * 16 + l16];
      float* op = out + (obase + dt * 16);
#pragma unroll
      for (int r = 0; r < 8; ++r)
        op[r * DD] = acc[r] * invr[r] + bv;
    }
  }
}

extern "C" void kernel_launch(void* const* d_in, const int* in_sizes, int n_in,
                              void* d_out, int out_size, void* d_ws, size_t ws_size,
                              hipStream_t stream) {
  (void)in_sizes; (void)n_in; (void)out_size; (void)d_ws; (void)ws_size;
  const float* src  = (const float*)d_in[0];
  const int*   adj  = (const int*)  d_in[1];
  const float* a0   = (const float*)d_in[2];
  const float* a1   = (const float*)d_in[3];
  const float* a2   = (const float*)d_in[4];
  const float* a3   = (const float*)d_in[5];
  const float* bias = (const float*)d_in[6];
  float* out = (float*)d_out;

  const size_t shmem = SMEM_BYTES;  // ~265 KB of the 320 KB WGP LDS
  LocalAggregator_gat_kernel<<<BATCH, 256, shmem, stream>>>(src, adj, a0, a1, a2, a3, bias, out);
}
